// TA_trans_3762391351592
// MI455X (gfx1250) — matrix operand order repaired and checked by reading, not compiled, not run on hardware
//
#include <hip/hip_runtime.h>

// ---------------------------------------------------------------------------
// CDNA5 (gfx1250) WMMA helpers: bf16 inputs, f32 accumulation.
// V_WMMA_F32_16X16X32_BF16, wave32, EXEC all-ones at every call site.
// ---------------------------------------------------------------------------
typedef __attribute__((ext_vector_type(16))) __bf16 v16bf;
typedef __attribute__((ext_vector_type(8)))  float  v8f;

__device__ __forceinline__ __bf16 f2bf(float f) {
  unsigned u = __builtin_bit_cast(unsigned, f);
  unsigned r = u + 0x7FFFu + ((u >> 16) & 1u);          // round-to-nearest-even
  unsigned short s = (unsigned short)(r >> 16);
  return __builtin_bit_cast(__bf16, s);
}

// A fragment: 16xK tile, rows m=0..15 at p[m*ld + k0 + k]
__device__ __forceinline__ v16bf load_a(const __bf16* p, int ld, int k0) {
  unsigned l = __lane_id();
  int hf = (l >> 4) & 1, m = l & 15;
  const __bf16* row = p + m * ld + k0;
  v16bf a;
#pragma unroll
  for (int i = 0; i < 8; ++i) {
    int k = hf * 16 + i * 2;
    a[2 * i]     = row[k];
    a[2 * i + 1] = row[k + 1];
  }
  return a;
}

// B fragment from row-major [k][n]: element(k,n) = p[(k0+k)*ld + n0+n]
__device__ __forceinline__ v16bf load_b(const __bf16* p, int ld, int k0, int n0) {
  unsigned l = __lane_id();
  int hf = (l >> 4) & 1, n = l & 15;
  const __bf16* col = p + (k0 + hf * 16) * ld + n0 + n;
  v16bf b;
#pragma unroll
  for (int i = 0; i < 8; ++i) {
    b[2 * i]     = col[(2 * i) * ld];
    b[2 * i + 1] = col[(2 * i + 1) * ld];
  }
  return b;
}

// B fragment from [n][k] storage (transposed): element(k,n) = p[(n0+n)*ld + k0+k]
__device__ __forceinline__ v16bf load_bT(const __bf16* p, int ld, int k0, int n0) {
  unsigned l = __lane_id();
  int hf = (l >> 4) & 1, n = l & 15;
  const __bf16* row = p + (n0 + n) * ld + k0 + hf * 16;
  v16bf b;
#pragma unroll
  for (int i = 0; i < 8; ++i) {
    b[2 * i]     = row[2 * i];
    b[2 * i + 1] = row[2 * i + 1];
  }
  return b;
}

__device__ __forceinline__ v8f wmma_bf16(v16bf a, v16bf b, v8f c) {
  return __builtin_amdgcn_wmma_f32_16x16x32_bf16(false, a, false, b, (short)0, c,
                                                 false, false);
}

// Problem constants
#define BATCH 4
#define CCH   64
#define HH    256
#define WW    256
#define HW    65536        // H*W
#define CHW   4194304      // C*H*W
#define C2HW  8388608      // 2C*H*W

// ---------------------------------------------------------------------------
// TDM (Tensor Data Mover) staging: one DMA per block pulls a [64][256] fp32
// tile (channel stride HW) from global into LDS.  Descriptor packing per
// CDNA5 ISA (2D tile, groups 2/3 zero).
// ---------------------------------------------------------------------------
#if __has_builtin(__builtin_amdgcn_tensor_load_to_lds)
#define HAVE_TDM 1
#else
#define HAVE_TDM 0
#endif

typedef unsigned tdm_v4u __attribute__((ext_vector_type(4)));
typedef int      tdm_v8i __attribute__((ext_vector_type(8)));
typedef int      tdm_v4i __attribute__((ext_vector_type(4)));

// Stage [nchan][256] fp32 tile; element (c,w) = gsrc[c*HW + w] -> lds_dst[c*256+w]
__device__ __forceinline__ void stage_rowtile_f32(float* lds_dst,
                                                  const float* gsrc, int nchan) {
#if HAVE_TDM
  if (threadIdx.x == 0) {
    unsigned long long ga = (unsigned long long)(const void*)gsrc;
    unsigned lds_off = (unsigned)(unsigned long long)(void*)lds_dst; // LDS byte addr
    tdm_v4u g0;
    g0[0] = 1u;                                            // count=1 user D#
    g0[1] = lds_off;                                       // lds_addr
    g0[2] = (unsigned)(ga & 0xffffffffu);                  // global_addr[31:0]
    g0[3] = (unsigned)((ga >> 32) & 0x01ffffffu) | (2u << 30); // addr[56:32]|type=2
    unsigned d0 = 256u, d1 = (unsigned)nchan, t0 = 256u, t1 = (unsigned)nchan;
    unsigned s0 = (unsigned)HW;                            // dim0 stride (elems)
    tdm_v8i g1;
    g1[0] = (int)(2u << 16);                               // data_size=4B, mask=0
    g1[1] = (int)((d0 & 0xffffu) << 16);                   // tensor_dim0 lo
    g1[2] = (int)((d0 >> 16) | ((d1 & 0xffffu) << 16));    // dim0 hi | dim1 lo
    g1[3] = (int)((d1 >> 16) | ((t0 & 0xffffu) << 16));    // dim1 hi | tile_dim0
    g1[4] = (int)(t1 & 0xffffu);                           // tile_dim1 (tile2=0)
    g1[5] = (int)s0;                                       // dim0_stride[31:0]
    g1[6] = 0;                                             // stride hi | dim1_stride
    g1[7] = 0;
    tdm_v4i z4 = {0, 0, 0, 0};
#if defined(__clang_major__) && (__clang_major__ >= 23)
    tdm_v8i z8 = {0, 0, 0, 0, 0, 0, 0, 0};
    __builtin_amdgcn_tensor_load_to_lds(g0, g1, z4, z4, z8, 0);
#else
    __builtin_amdgcn_tensor_load_to_lds(g0, g1, z4, z4, 0);
#endif
    __builtin_amdgcn_s_wait_tensorcnt(0);
  }
#else
  for (int c = 0; c < nchan; ++c)
    lds_dst[c * 256 + threadIdx.x] = gsrc[c * HW + threadIdx.x];
#endif
}

// ---------------------------------------------------------------------------
// K1: v_img = pv(x) (WMMA 1x1 conv) fused with t = mean(leaky(tac(v_img)))
// One block per image row (256 pixels).  LDS: xb(32K) wb(8K) xf|vv(64K)
// ---------------------------------------------------------------------------
__global__ void k_pv_tac(const float* __restrict__ x, const float* __restrict__ pvw,
                         const float* __restrict__ pvb, const float* __restrict__ tacw,
                         const float* __restrict__ tacb,
                         float* __restrict__ vimg, float* __restrict__ timg) {
  extern __shared__ char smem[];
  __bf16* xb = (__bf16*)smem;             // [256][64]
  __bf16* wb = (__bf16*)(smem + 32768);   // [c=64][d=64]  (pv_w transposed)
  float*  xf = (float*)(smem + 40960);    // fp32 staging [64][256]
  float*  vv = (float*)(smem + 40960);    // [256][64]  (reuses xf region)

  const int tid = threadIdx.x;
  const int b = blockIdx.x >> 8, h = blockIdx.x & 255;
  stage_rowtile_f32(xf, x + (size_t)b * CHW + h * WW, 64);   // TDM
  for (int i = tid; i < 4096; i += 256) {
    int d = i >> 6, c = i & 63;
    wb[c * 64 + d] = f2bf(pvw[i]);
  }
  __syncthreads();
  for (int c = 0; c < 64; ++c) xb[tid * 64 + c] = f2bf(xf[c * 256 + tid]);
  __syncthreads();   // xf consumed; region becomes vv

  const int wave = tid >> 5;
  const unsigned l = __lane_id();
  const int hf = (l >> 4) & 1, ln = l & 15;
  for (int it = 0; it < 8; ++it) {         // uniform trip count (no EXEC churn)
    int j = it * 8 + wave;                 // 16 M-tiles x 4 N-tiles
    int mt = j >> 2, nt = j & 3;
    v8f acc = {};
    acc = wmma_bf16(load_a(xb + mt * 16 * 64, 64, 0),  load_b(wb, 64, 0,  nt * 16), acc);
    acc = wmma_bf16(load_a(xb + mt * 16 * 64, 64, 32), load_b(wb, 64, 32, nt * 16), acc);
#pragma unroll
    for (int r = 0; r < 8; ++r) {
      int row = mt * 16 + r + 8 * hf, col = nt * 16 + ln;
      vv[row * 64 + col] = acc[r] + pvb[col];
    }
  }
  __syncthreads();

  // tac 64->16, leaky-relu, mean over 16 channels
  float tv = 0.f;
  for (int j = 0; j < 16; ++j) {
    float s = tacb[j];
    for (int c = 0; c < 64; ++c) s = fmaf(vv[tid * 64 + c], tacw[j * 64 + c], s);
    tv += (s > 0.f) ? s : 0.1f * s;
  }
  timg[(size_t)b * HW + h * WW + tid] = tv * (1.f / 16.f);
  float* vout = vimg + (size_t)b * CHW + h * WW + tid;
  for (int c = 0; c < 64; ++c) vout[c * HW] = vv[tid * 64 + c];
}

// ---------------------------------------------------------------------------
// K2: per-window variance (ddof=1) -> straight-through binary mask
// ---------------------------------------------------------------------------
__global__ void k_mask(const float* __restrict__ timg, const float* __restrict__ thr,
                       float* __restrict__ mask) {
  __shared__ float s1[256], s2[256];
  const int tid = threadIdx.x, win = blockIdx.x;
  const int b = win >> 8, n = win & 255;
  const int h = (n >> 4) * 16 + (tid >> 4), w = (n & 15) * 16 + (tid & 15);
  float v = timg[(size_t)b * HW + h * WW + w];
  s1[tid] = v; s2[tid] = v * v;
  __syncthreads();
  for (int s = 128; s > 0; s >>= 1) {
    if (tid < s) { s1[tid] += s1[tid + s]; s2[tid] += s2[tid + s]; }
    __syncthreads();
  }
  if (tid == 0) {
    float mean = s1[0] * (1.f / 256.f);
    float var  = (s2[0] - 256.f * mean * mean) * (1.f / 255.f);
    float soft = 1.f - 1.f / (1.f + __expf(-(10.f * (thr[0] - var))));
    mask[win] = (soft > 0.5f) ? 1.f : 0.f;
  }
}

// ---------------------------------------------------------------------------
// K3: windowed attention.  One block per window (T=256, C=64), 8 waves.
// LDS 160KB: v1(32K) q(32K) k(32K) stage(64K: xw+weights, later softmax P)
// ---------------------------------------------------------------------------
__global__ void k_attn(const float* __restrict__ x, const float* __restrict__ vimg,
                       const float* __restrict__ mask,
                       const float* __restrict__ qw, const float* __restrict__ qbias,
                       const float* __restrict__ kw, const float* __restrict__ kbias,
                       float* __restrict__ fimg) {
  extern __shared__ char smem[];
  __bf16* v1b = (__bf16*)smem;                     // [256][64]
  __bf16* qb  = (__bf16*)(smem + 32768);           // [256][64]
  __bf16* kb  = (__bf16*)(smem + 65536);           // [256][64]
  __bf16* stg = (__bf16*)(smem + 98304);           // 64KB region
  __bf16* xw  = stg;                               // [256][64]
  __bf16* wq  = (__bf16*)(smem + 98304 + 32768);   // [c][d]
  __bf16* wk  = (__bf16*)(smem + 98304 + 40960);   // [c][d]

  const int tid = threadIdx.x;
  const int win = blockIdx.x;
  const int b = win >> 8, n = win & 255;
  const int h0 = (n >> 4) * 16, w0 = (n & 15) * 16;
  const float m = mask[win];

  {
    const int t = tid;
    const int hh = h0 + (t >> 4), ww = w0 + (t & 15);
    const float* xp = x    + (size_t)b * CHW + hh * WW + ww;
    const float* vp = vimg + (size_t)b * CHW + hh * WW + ww;
    for (int c = 0; c < 64; ++c) {
      xw[t * 64 + c]  = f2bf(xp[c * HW] * m);
      v1b[t * 64 + c] = f2bf(vp[c * HW] * m);
    }
  }
  for (int i = tid; i < 4096; i += 256) {
    int d = i >> 6, c = i & 63;
    wq[c * 64 + d] = f2bf(qw[i]);
    wk[c * 64 + d] = f2bf(kw[i]);
  }
  __syncthreads();

  const int wave = tid >> 5;
  const unsigned l = __lane_id();
  const int hf = (l >> 4) & 1, ln = l & 15;

  // q / k projections (WMMA), bf16 results to LDS
  for (int mi = 0; mi < 2; ++mi) {
    int mt = wave * 2 + mi;
    for (int nt = 0; nt < 4; ++nt) {
      v8f aq = {}, ak = {};
      v16bf a0 = load_a(xw + mt * 16 * 64, 64, 0);
      v16bf a1 = load_a(xw + mt * 16 * 64, 64, 32);
      aq = wmma_bf16(a0, load_b(wq, 64, 0,  nt * 16), aq);
      aq = wmma_bf16(a1, load_b(wq, 64, 32, nt * 16), aq);
      ak = wmma_bf16(a0, load_b(wk, 64, 0,  nt * 16), ak);
      ak = wmma_bf16(a1, load_b(wk, 64, 32, nt * 16), ak);
#pragma unroll
      for (int r = 0; r < 8; ++r) {
        int row = mt * 16 + r + 8 * hf, col = nt * 16 + ln;
        qb[row * 64 + col] = f2bf(aq[r] + qbias[col]);
        kb[row * 64 + col] = f2bf(ak[r] + kbias[col]);
      }
    }
  }
  __syncthreads();   // xw/wq/wk dead from here; stage reused for P

  __bf16* P = stg + wave * 4096;   // per-wave 16x256 bf16 softmax tile
  for (int mi = 0; mi < 2; ++mi) {
    int mt = wave * 2 + mi;
    v8f S[16] = {};
    v16bf aq0 = load_a(qb + mt * 16 * 64, 64, 0);
    v16bf aq1 = load_a(qb + mt * 16 * 64, 64, 32);
#pragma unroll
    for (int nt = 0; nt < 16; ++nt) {    // S = q . k^T  (16x256 strip)
      S[nt] = wmma_bf16(aq0, load_bT(kb, 64, 0,  nt * 16), S[nt]);
      S[nt] = wmma_bf16(aq1, load_bT(kb, 64, 32, nt * 16), S[nt]);
    }
    // row softmax: C-layout row r+8*hf lives on the 16 lanes of this half
#pragma unroll
    for (int r = 0; r < 8; ++r) {
      float mx = S[0][r];
#pragma unroll
      for (int nt = 1; nt < 16; ++nt) mx = fmaxf(mx, S[nt][r]);
      for (int d = 1; d < 16; d <<= 1) mx = fmaxf(mx, __shfl_xor(mx, d));
      float sum = 0.f;
#pragma unroll
      for (int nt = 0; nt < 16; ++nt) {
        float e = __expf(S[nt][r] - mx);
        S[nt][r] = e; sum += e;
      }
      for (int d = 1; d < 16; d <<= 1) sum += __shfl_xor(sum, d);
      float inv = 1.f / sum;
#pragma unroll
      for (int nt = 0; nt < 16; ++nt) S[nt][r] *= inv;
    }
    // stage P (same-wave LDS is in-order: no barrier needed)
#pragma unroll
    for (int nt = 0; nt < 16; ++nt)
#pragma unroll
      for (int r = 0; r < 8; ++r)
        P[(r + 8 * hf) * 256 + nt * 16 + ln] = f2bf(S[nt][r]);
    // f = P . v1   (K = 256)
    for (int nt2 = 0; nt2 < 4; ++nt2) {
      v8f f = {};
#pragma unroll
      for (int kt = 0; kt < 8; ++kt)
        f = wmma_bf16(load_a(P, 256, kt * 32), load_b(v1b, 64, kt * 32, nt2 * 16), f);
#pragma unroll
      for (int r = 0; r < 8; ++r) {
        int t2 = mt * 16 + r + 8 * hf, c = nt2 * 16 + ln;
        fimg[(size_t)b * CHW + c * HW + (h0 + (t2 >> 4)) * WW + (w0 + (t2 & 15))] = f[r];
      }
    }
  }
}

// ---------------------------------------------------------------------------
// K4: depthwise 3x3 pad1 on v2 = v_img*(1-m)   (mask applied per-tap window)
// ---------------------------------------------------------------------------
__global__ void k_dw1(const float* __restrict__ vimg, const float* __restrict__ mask,
                      const float* __restrict__ w, const float* __restrict__ bias,
                      float* __restrict__ out) {
  const int tid = threadIdx.x;
  const int bc = blockIdx.x >> 8, h = blockIdx.x & 255;
  const int b = bc >> 6, c = bc & 63;
  float acc = bias[c];
  __builtin_prefetch(&vimg[(size_t)bc * HW + (h < 254 ? h + 2 : h) * WW + tid], 0, 0);
  for (int ky = 0; ky < 3; ++ky) {
    int hh = h + ky - 1; if (hh < 0 || hh > 255) continue;
    for (int kx = 0; kx < 3; ++kx) {
      int ww = tid + kx - 1; if (ww < 0 || ww > 255) continue;
      float mm = mask[b * 256 + (hh >> 4) * 16 + (ww >> 4)];
      acc = fmaf(w[c * 9 + ky * 3 + kx],
                 vimg[(size_t)bc * HW + hh * WW + ww] * (1.f - mm), acc);
    }
  }
  out[(size_t)bc * HW + h * WW + tid] = acc;
}

// ---------------------------------------------------------------------------
// K5: depthwise 3x3 dilation-2 pad2; y = f_img + result (in place on y)
// ---------------------------------------------------------------------------
__global__ void k_dw2(const float* __restrict__ tmp, const float* __restrict__ w,
                      const float* __restrict__ bias, float* __restrict__ y) {
  const int tid = threadIdx.x;
  const int bc = blockIdx.x >> 8, h = blockIdx.x & 255;
  const int c = bc & 63;
  float acc = bias[c];
  for (int ky = 0; ky < 3; ++ky) {
    int hh = h + 2 * (ky - 1); if (hh < 0 || hh > 255) continue;
    for (int kx = 0; kx < 3; ++kx) {
      int ww = tid + 2 * (kx - 1); if (ww < 0 || ww > 255) continue;
      acc = fmaf(w[c * 9 + ky * 3 + kx], tmp[(size_t)bc * HW + hh * WW + ww], acc);
    }
  }
  size_t idx = (size_t)bc * HW + h * WW + tid;
  y[idx] = y[idx] + acc;
}

// ---------------------------------------------------------------------------
// K6/K9: z = WMMA_1x1(src, W) (+bias) + res, then channel LayerNorm.
// LDS: sb(32K) wb(8K) sf|vv(64K)
// ---------------------------------------------------------------------------
__global__ void k_proj_ln(const float* __restrict__ src, const float* __restrict__ W,
                          const float* __restrict__ bias, const float* __restrict__ res,
                          const float* __restrict__ gamma, const float* __restrict__ beta,
                          float* __restrict__ out) {
  extern __shared__ char smem[];
  __bf16* sb = (__bf16*)smem;             // [256][64]
  __bf16* wb = (__bf16*)(smem + 32768);   // [c][d]
  float*  sf = (float*)(smem + 40960);    // fp32 staging [64][256]
  float*  vv = (float*)(smem + 40960);    // [256][64]

  const int tid = threadIdx.x;
  const int b = blockIdx.x >> 8, h = blockIdx.x & 255;
  stage_rowtile_f32(sf, src + (size_t)b * CHW + h * WW, 64);   // TDM
  for (int i = tid; i < 4096; i += 256) {
    int d = i >> 6, c = i & 63;
    wb[c * 64 + d] = f2bf(W[i]);
  }
  __syncthreads();
  for (int c = 0; c < 64; ++c) sb[tid * 64 + c] = f2bf(sf[c * 256 + tid]);
  __syncthreads();   // sf consumed; region becomes vv

  const int wave = tid >> 5;
  const unsigned l = __lane_id();
  const int hf = (l >> 4) & 1, ln = l & 15;
  for (int it = 0; it < 8; ++it) {
    int j = it * 8 + wave;
    int mt = j >> 2, nt = j & 3;
    v8f acc = {};
    acc = wmma_bf16(load_a(sb + mt * 16 * 64, 64, 0),  load_b(wb, 64, 0,  nt * 16), acc);
    acc = wmma_bf16(load_a(sb + mt * 16 * 64, 64, 32), load_b(wb, 64, 32, nt * 16), acc);
#pragma unroll
    for (int r = 0; r < 8; ++r) {
      int row = mt * 16 + r + 8 * hf, col = nt * 16 + ln;
      vv[row * 64 + col] = acc[r] + (bias ? bias[col] : 0.f);
    }
  }
  __syncthreads();

  const float* rrow = res + (size_t)b * CHW + h * WW + tid;
  float sum = 0.f, sq = 0.f;
  for (int c = 0; c < 64; ++c) {
    float o = vv[tid * 64 + c] + rrow[c * HW];
    vv[tid * 64 + c] = o;
    sum += o; sq += o * o;
  }
  float mu = sum * (1.f / 64.f);
  float var = sq * (1.f / 64.f) - mu * mu;
  float rs = rsqrtf(var + 1e-5f);
  float* orow = out + (size_t)b * CHW + h * WW + tid;
  for (int c = 0; c < 64; ++c)
    orow[c * HW] = (vv[tid * 64 + c] - mu) * rs * gamma[c] + beta[c];
}

// ---------------------------------------------------------------------------
// K7: 1x1 conv C -> 2C (ffn_in), WMMA, N=128.  LDS: sb(32K) wb(16K) sf|vv(128K)
// ---------------------------------------------------------------------------
__global__ void k_ffn_in(const float* __restrict__ x1, const float* __restrict__ W,
                         float* __restrict__ h0) {
  extern __shared__ char smem[];
  __bf16* sb = (__bf16*)smem;             // [256][64]
  __bf16* wb = (__bf16*)(smem + 32768);   // [c=64][d=128]
  float*  sf = (float*)(smem + 49152);    // fp32 staging [64][256]
  float*  vv = (float*)(smem + 49152);    // [256][128]

  const int tid = threadIdx.x;
  const int b = blockIdx.x >> 8, h = blockIdx.x & 255;
  stage_rowtile_f32(sf, x1 + (size_t)b * CHW + h * WW, 64);   // TDM
  for (int i = tid; i < 8192; i += 256) {
    int d = i >> 6, c = i & 63;
    wb[c * 128 + d] = f2bf(W[i]);
  }
  __syncthreads();
  for (int c = 0; c < 64; ++c) sb[tid * 64 + c] = f2bf(sf[c * 256 + tid]);
  __syncthreads();   // sf consumed; region becomes vv

  const int wave = tid >> 5;
  const unsigned l = __lane_id();
  const int hf = (l >> 4) & 1, ln = l & 15;
  for (int it = 0; it < 16; ++it) {        // 16 M-tiles x 8 N-tiles
    int j = it * 8 + wave;
    int mt = j >> 3, nt = j & 7;
    v8f acc = {};
    acc = wmma_bf16(load_a(sb + mt * 16 * 64, 64, 0),  load_b(wb, 128, 0,  nt * 16), acc);
    acc = wmma_bf16(load_a(sb + mt * 16 * 64, 64, 32), load_b(wb, 128, 32, nt * 16), acc);
#pragma unroll
    for (int r = 0; r < 8; ++r) {
      int row = mt * 16 + r + 8 * hf, col = nt * 16 + ln;
      vv[row * 128 + col] = acc[r];
    }
  }
  __syncthreads();

  float* orow = h0 + (size_t)b * C2HW + h * WW + tid;
  for (int c = 0; c < 128; ++c) orow[c * HW] = vv[tid * 128 + c];
}

// ---------------------------------------------------------------------------
// K8: depthwise 3x3 pad1 on 2C channels + exact-GELU gate: g = gelu(h1)*h2
// ---------------------------------------------------------------------------
__global__ void k_dw_gate(const float* __restrict__ h0, const float* __restrict__ w,
                          float* __restrict__ g) {
  const int tid = threadIdx.x;
  const int bc = blockIdx.x >> 8, h = blockIdx.x & 255;
  const int b = bc >> 6, c = bc & 63;
  const float* p1 = h0 + (size_t)b * C2HW + (size_t)c * HW;
  const float* p2 = p1 + (size_t)64 * HW;
  float a = 0.f, bb = 0.f;
  for (int ky = 0; ky < 3; ++ky) {
    int hh = h + ky - 1; if (hh < 0 || hh > 255) continue;
    for (int kx = 0; kx < 3; ++kx) {
      int ww = tid + kx - 1; if (ww < 0 || ww > 255) continue;
      int off = hh * WW + ww;
      a  = fmaf(w[c * 9 + ky * 3 + kx],        p1[off], a);
      bb = fmaf(w[(c + 64) * 9 + ky * 3 + kx], p2[off], bb);
    }
  }
  float gl = a * 0.5f * (1.f + erff(a * 0.70710678f));
  g[(size_t)bc * HW + h * WW + tid] = gl * bb;
}

// ---------------------------------------------------------------------------
// Launcher
// ---------------------------------------------------------------------------
extern "C" void kernel_launch(void* const* d_in, const int* in_sizes, int n_in,
                              void* d_out, int out_size, void* d_ws, size_t ws_size,
                              hipStream_t stream) {
  (void)in_sizes; (void)n_in; (void)out_size; (void)ws_size;
  const float* x     = (const float*)d_in[0];
  const float* pvw   = (const float*)d_in[1];
  const float* pvb   = (const float*)d_in[2];
  const float* tacw  = (const float*)d_in[3];
  const float* tacb  = (const float*)d_in[4];
  const float* thr   = (const float*)d_in[5];
  const float* qw    = (const float*)d_in[6];
  const float* qbias = (const float*)d_in[7];
  const float* kw    = (const float*)d_in[8];
  const float* kbias = (const float*)d_in[9];
  const float* c1w   = (const float*)d_in[10];
  const float* c1b   = (const float*)d_in[11];
  const float* c2w   = (const float*)d_in[12];
  const float* c2b   = (const float*)d_in[13];
  const float* pow_  = (const float*)d_in[14];
  const float* pob   = (const float*)d_in[15];
  const float* n1g   = (const float*)d_in[16];
  const float* n1b   = (const float*)d_in[17];
  const float* n2g   = (const float*)d_in[18];
  const float* n2b   = (const float*)d_in[19];
  const float* fiw   = (const float*)d_in[20];
  const float* fdw   = (const float*)d_in[21];
  const float* fow   = (const float*)d_in[22];

  char* ws = (char*)d_ws;
  const size_t SZ = (size_t)BATCH * CCH * HH * WW * sizeof(float);  // 64 MB
  size_t o = 0;
  float* vimg = (float*)(ws + o); o += SZ;       // pv output (image layout)
  float* y    = (float*)(ws + o); o += SZ;       // f_img, then f_img + v2c
  float* tmp  = (float*)(ws + o); o += SZ;       // depthwise intermediate
  float* x1   = (float*)(ws + o); o += SZ;       // after first residual+LN
  float* h0   = (float*)(ws + o); o += 2 * SZ;   // ffn hidden (2C)
  float* timg = (float*)(ws + o); o += SZ / 64;  // t image (1 channel)
  float* mask = (float*)(ws + o); o += 4096;     // per-window binary mask
  float* g    = vimg;                            // reuse: v_img dead after K4

  dim3 blk(256);
  k_pv_tac <<<dim3(BATCH * HH),       blk, 106496, stream>>>(x, pvw, pvb, tacw, tacb, vimg, timg);
  k_mask   <<<dim3(BATCH * 256),      blk, 0,      stream>>>(timg, thr, mask);
  k_attn   <<<dim3(BATCH * 256),      blk, 163840, stream>>>(x, vimg, mask, qw, qbias, kw, kbias, y);
  k_dw1    <<<dim3(BATCH * CCH * HH), blk, 0,      stream>>>(vimg, mask, c1w, c1b, tmp);
  k_dw2    <<<dim3(BATCH * CCH * HH), blk, 0,      stream>>>(tmp, c2w, c2b, y);
  k_proj_ln<<<dim3(BATCH * HH),       blk, 106496, stream>>>(y, pow_, pob, x, n1g, n1b, x1);
  k_ffn_in <<<dim3(BATCH * HH),       blk, 180224, stream>>>(x1, fiw, h0);
  k_dw_gate<<<dim3(BATCH * CCH * HH), blk, 0,      stream>>>(h0, fdw, g);
  k_proj_ln<<<dim3(BATCH * HH),       blk, 106496, stream>>>(g, fow, nullptr, x1, n2g, n2b,
                                                             (float*)d_out);
}
